// SpatialTransformer_19258633355617
// MI455X (gfx1250) — matrix operand order repaired and checked by reading, not compiled, not run on hardware
//
#include <hip/hip_runtime.h>
#include <math.h>

// ---------------------------------------------------------------------------
// SpatialTransformer block for MI455X (gfx1250), wave32 + WMMA f16->f32.
// f16 activation dataflow; LDS staging via global_load_async_to_lds_b128.
// ---------------------------------------------------------------------------

typedef __attribute__((ext_vector_type(16))) _Float16 v16h;
typedef __attribute__((ext_vector_type(8)))  _Float16 v8h;
typedef __attribute__((ext_vector_type(8)))  float    v8f;

union AFrag { v16h v; v8h p[2]; _Float16 h[16]; };
union CFrag { v8f  v; float    f[8];  };
union H8    { v8h  v; _Float16 h[8];  };

#define NTOK 4096   // H*W
#define CCH  320    // channels
#define NHEAD 8
#define HDIM  40

// Async copy of 16 bytes global -> LDS (CDNA5 GLOBAL_LOAD_ASYNC_TO_LDS_B128,
// tracked by ASYNCcnt). lds must be a pointer into __shared__ memory.
__device__ __forceinline__ void async_g2l_b128(void* lds, const void* g) {
    unsigned l = (unsigned)(uintptr_t)lds;                 // LDS byte offset
    unsigned long long a = (unsigned long long)(uintptr_t)g;
    asm volatile("global_load_async_to_lds_b128 %0, %1, off"
                 :: "v"(l), "v"(a) : "memory");
}
__device__ __forceinline__ void wait_async() {
    asm volatile("s_wait_asynccnt 0x0" ::: "memory");
}

// Convert 8 contiguous f32 (two float4 loads) to 8 f16
__device__ __forceinline__ H8 cvt8(const float* p) {
    const float4* p4 = (const float4*)p;
    float4 u0 = p4[0], u1 = p4[1];
    H8 r;
    r.h[0] = (_Float16)u0.x; r.h[1] = (_Float16)u0.y;
    r.h[2] = (_Float16)u0.z; r.h[3] = (_Float16)u0.w;
    r.h[4] = (_Float16)u1.x; r.h[5] = (_Float16)u1.y;
    r.h[6] = (_Float16)u1.z; r.h[7] = (_Float16)u1.w;
    return r;
}
__device__ __forceinline__ H8 zero8() {
    H8 r;
    for (int j = 0; j < 8; j++) r.h[j] = (_Float16)0.f;
    return r;
}

// A fragment (16x32 f16): the 16 halves of lane-group g are one contiguous
// run: K in [16g, 16g+16)  -> two b128 LDS loads.
__device__ __forceinline__ v16h load_a_frag(const _Float16* rowp, int g) {
    AFrag a;
    a.p[0] = *(const v8h*)(rowp + 16 * g);
    a.p[1] = *(const v8h*)(rowp + 16 * g + 8);
    return a.v;
}
// B fragment when the 16 halves (K = 16g .. 16g+15) are contiguous in LDS.
__device__ __forceinline__ v16h load_b_frag(const _Float16* p) {
    AFrag b;
    b.p[0] = *(const v8h*)(p);
    b.p[1] = *(const v8h*)(p + 8);
    return b.v;
}

// ---------------------------------------------------------------------------
// f32 -> f16 bulk convert (for weights), 8 elements per thread
// ---------------------------------------------------------------------------
__global__ __launch_bounds__(256) void cvt_f16_kernel(
    const float* __restrict__ src, _Float16* __restrict__ dst, int n8) {
    int i = blockIdx.x * 256 + threadIdx.x;
    if (i < n8) {
        H8 h = cvt8(src + (size_t)i * 8);
        *(v8h*)(dst + (size_t)i * 8) = h.v;
    }
}

// ---------------------------------------------------------------------------
// GroupNorm(32 groups) over x[C][N], writes transposed out[N][C] (f16)
// ---------------------------------------------------------------------------
__global__ __launch_bounds__(256) void groupnorm_kernel(
    const float* __restrict__ x, const float* __restrict__ w,
    const float* __restrict__ b, _Float16* __restrict__ out) {
    __shared__ float ssum[256], ssq[256];
    int g = blockIdx.x;           // 32 groups
    int t = threadIdx.x;
    const int CPG = CCH / 32;     // 10 channels per group
    const int CNT = CPG * NTOK;   // 40960
    float s = 0.f, q = 0.f;
    for (int i = t; i < CNT; i += 256) {
        int ci = i / NTOK, n = i % NTOK;
        float v = x[(g * CPG + ci) * NTOK + n];
        s += v; q += v * v;
    }
    ssum[t] = s; ssq[t] = q; __syncthreads();
    for (int off = 128; off > 0; off >>= 1) {
        if (t < off) { ssum[t] += ssum[t + off]; ssq[t] += ssq[t + off]; }
        __syncthreads();
    }
    float mean = ssum[0] / (float)CNT;
    float var  = ssq[0] / (float)CNT - mean * mean;
    float rstd = rsqrtf(var + 1e-5f);
    for (int i = t; i < CNT; i += 256) {
        int ci = i / NTOK, n = i % NTOK;
        int c = g * CPG + ci;
        float v = x[c * NTOK + n];
        out[(size_t)n * CCH + c] = (_Float16)((v - mean) * rstd * w[c] + b[c]);
    }
}

// ---------------------------------------------------------------------------
// LayerNorm over channel dim of x[C][N] per token, writes out[N][C] (f16).
// One wave32 per token.
// ---------------------------------------------------------------------------
__global__ __launch_bounds__(256) void layernorm_kernel(
    const float* __restrict__ x, const float* __restrict__ w,
    const float* __restrict__ b, _Float16* __restrict__ out) {
    int wave = threadIdx.x >> 5, lane = threadIdx.x & 31;
    int n = blockIdx.x * 8 + wave;
    float s = 0.f, q = 0.f;
    for (int c = lane; c < CCH; c += 32) {
        float v = x[(size_t)c * NTOK + n]; s += v; q += v * v;
    }
    for (int m = 16; m > 0; m >>= 1) {
        s += __shfl_xor(s, m, 32);
        q += __shfl_xor(q, m, 32);
    }
    float mean = s / (float)CCH;
    float var  = q / (float)CCH - mean * mean;
    float rstd = rsqrtf(var + 1e-5f);
    for (int c = lane; c < CCH; c += 32) {
        float v = x[(size_t)c * NTOK + n];
        out[(size_t)n * CCH + c] = (_Float16)((v - mean) * rstd * w[c] + b[c]);
    }
}

// ---------------------------------------------------------------------------
// WMMA GEMM: C[M][O] = A[M][K] * W[O][K]^T + bias[O], optional exact GELU.
// Block tile 64x32, 8 waves (4x2), each wave one 16x16 WMMA tile, K step 32.
// A and W are f16 in memory; LDS staging is async b128 copies (no VALU).
// Writes f32 and/or f16 outputs (either pointer may be null).
// ---------------------------------------------------------------------------
__global__ __launch_bounds__(256) void gemm_kernel(
    const _Float16* __restrict__ A, const _Float16* __restrict__ W,
    const float* __restrict__ bias, float* __restrict__ C32,
    _Float16* __restrict__ C16, int M, int K, int O, int act) {
    __shared__ _Float16 As[64][32];   // [m_local][k]
    __shared__ _Float16 Bs[32][32];   // [o_local][k]  (= W row-major slice)
    int t = threadIdx.x;
    int wave = t >> 5, lane = t & 31, g = lane >> 4, l16 = lane & 15;
    int wm = wave >> 1, wn = wave & 1;
    int bm = blockIdx.x, bo = blockIdx.y;
    int arow = t >> 2, ac0 = (t & 3) * 8;        // A staging map (64 x 32)
    int am = bm * 64 + arow;
    bool avalid = (am < M);
    const _Float16* aptr = A + (size_t)am * K + ac0;
    int ol = t >> 2, ok0 = (t & 3) * 8;          // B staging map (32 x 32)
    const _Float16* wptr = W + (size_t)(bo * 32 + (ol & 31)) * K + ok0;

    if (!avalid) *(v8h*)&As[arow][ac0] = zero8().v;  // stays zero for all kk

    CFrag acc;
    for (int r = 0; r < 8; r++) acc.f[r] = 0.f;

    for (int kk = 0; kk < K; kk += 32) {
        if (avalid) async_g2l_b128(&As[arow][ac0], aptr + kk);
        if (t < 128) async_g2l_b128(&Bs[ol][ok0], wptr + kk);
        if (kk + 32 < K && avalid)
            __builtin_prefetch(aptr + kk + 32, 0, 3);
        wait_async();
        __syncthreads();
        v16h a = load_a_frag(&As[wm * 16 + l16][0], g);
        v16h b = load_b_frag(&Bs[wn * 16 + l16][16 * g]);
        acc.v = __builtin_amdgcn_wmma_f32_16x16x32_f16(
            false, a, false, b, (short)0, acc.v, false, false);
        __syncthreads();
    }
    for (int r = 0; r < 8; r++) {
        int row = bm * 64 + wm * 16 + 8 * g + r;
        int col = bo * 32 + wn * 16 + l16;
        if (row < M) {
            float v = acc.f[r] + bias[col];
            if (act) v = 0.5f * v * (1.f + erff(v * 0.70710678f));  // exact GELU
            if (C32) C32[(size_t)row * O + col] = v;
            if (C16) C16[(size_t)row * O + col] = (_Float16)v;
        }
    }
}

// ---------------------------------------------------------------------------
// Flash attention, one head / 64-query tile per block, 64-key blocks with
// online softmax. Head dim 40 zero-padded to 64 (whole 8-chunks). Q/K/V f16
// in memory, staged to LDS with async b128 copies. Out f16.
// ---------------------------------------------------------------------------
__global__ __launch_bounds__(256) void flash_attn_kernel(
    const _Float16* __restrict__ Q, int ldq,
    const _Float16* __restrict__ Kp, int ldk,
    const _Float16* __restrict__ Vp, int ldv,
    _Float16* __restrict__ Out, int ldo, int Lk, float scale) {
    __shared__ _Float16 Qs[64][64];   // [query][d]
    __shared__ _Float16 Ks[64][64];   // [key][d]
    __shared__ _Float16 Vs[64][64];   // [key][d]
    __shared__ float    Sf[64][64];
    __shared__ _Float16 Pf[64][64];
    __shared__ float rowm[64], rowl[64], rowa[64];

    int t = threadIdx.x;
    int wave = t >> 5, lane = t & 31, g = lane >> 4, l16 = lane & 15;
    int wm = wave >> 1, wn = wave & 1;
    int h = blockIdx.y;
    int coff = h * HDIM;
    int q0 = blockIdx.x * 64;
    int srow = t >> 3;                // 0..31, staging row (2 passes)
    int sd0 = (t & 7) * 8;            // 0..56, staging d-chunk
    bool dvalid = (sd0 < HDIM);       // whole chunk in/out (40 % 8 == 0)

    // zero-fill the d-padding chunks once (async copies never touch them)
    for (int pass = 0; pass < 2; pass++) {
        int r = srow + pass * 32;
        if (dvalid) {
            async_g2l_b128(&Qs[r][sd0], Q + (size_t)(q0 + r) * ldq + coff + sd0);
        } else {
            H8 z = zero8();
            *(v8h*)&Qs[r][sd0] = z.v;
            *(v8h*)&Ks[r][sd0] = z.v;
            *(v8h*)&Vs[r][sd0] = z.v;
        }
    }
    if (t < 64) { rowm[t] = -3.0e38f; rowl[t] = 0.f; }
    CFrag acc[2];
    for (int ct = 0; ct < 2; ct++)
        for (int r = 0; r < 8; r++) acc[ct].f[r] = 0.f;

    int nkb = (Lk + 63) / 64;
    for (int kb = 0; kb < nkb; kb++) {
        for (int pass = 0; pass < 2; pass++) {    // stage K and V (8-chunks)
            int kc = srow + pass * 32;
            int key = kb * 64 + kc;
            if (dvalid) {
                if (key < Lk) {
                    async_g2l_b128(&Ks[kc][sd0], Kp + (size_t)key * ldk + coff + sd0);
                    async_g2l_b128(&Vs[kc][sd0], Vp + (size_t)key * ldv + coff + sd0);
                } else {
                    H8 z = zero8();
                    *(v8h*)&Ks[kc][sd0] = z.v;
                    *(v8h*)&Vs[kc][sd0] = z.v;
                }
            }
        }
        wait_async();
        __syncthreads();
        // S = Q K^T * scale  (each wave: 16 rows x 32 key-cols)
        for (int ct = 0; ct < 2; ct++) {
            CFrag s;
            for (int r = 0; r < 8; r++) s.f[r] = 0.f;
            int keycol = wn * 32 + ct * 16 + l16;
            for (int ks = 0; ks < 64; ks += 32) {
                v16h a = load_a_frag(&Qs[wm * 16 + l16][ks], g);
                v16h b = load_b_frag(&Ks[keycol][ks + 16 * g]);
                s.v = __builtin_amdgcn_wmma_f32_16x16x32_f16(
                    false, a, false, b, (short)0, s.v, false, false);
            }
            for (int r = 0; r < 8; r++)
                Sf[wm * 16 + 8 * g + r][keycol] = s.f[r] * scale;
        }
        __syncthreads();
        // online softmax, one thread per query row
        if (t < 64) {
            int row = t;
            int kmax = Lk - kb * 64; if (kmax > 64) kmax = 64;
            float mt = -3.0e38f;
            for (int c = 0; c < kmax; c++) mt = fmaxf(mt, Sf[row][c]);
            float mold = rowm[row];
            float mnew = fmaxf(mold, mt);
            float alpha = expf(mold - mnew);
            float sum = 0.f;
            for (int c = 0; c < 64; c++) {
                float p = (c < kmax) ? expf(Sf[row][c] - mnew) : 0.f;
                Pf[row][c] = (_Float16)p;
                sum += p;
            }
            rowl[row] = rowl[row] * alpha + sum;
            rowm[row] = mnew;
            rowa[row] = alpha;
        }
        __syncthreads();
        // O = O*alpha + P @ V
        for (int ct = 0; ct < 2; ct++) {
            for (int r = 0; r < 8; r++) acc[ct].f[r] *= rowa[wm * 16 + 8 * g + r];
            int dcol = wn * 32 + ct * 16 + l16;
            for (int ks = 0; ks < 64; ks += 32) {
                v16h a = load_a_frag(&Pf[wm * 16 + l16][ks], g);
                AFrag bf;   // B[k=key][n=d] from Vs[key][d]: strided column read
                for (int i = 0; i < 16; i++) bf.h[i] = Vs[ks + 16 * g + i][dcol];
                acc[ct].v = __builtin_amdgcn_wmma_f32_16x16x32_f16(
                    false, a, false, bf.v, (short)0, acc[ct].v, false, false);
            }
        }
        __syncthreads();
    }
    for (int ct = 0; ct < 2; ct++) {
        for (int r = 0; r < 8; r++) {
            int rl = wm * 16 + 8 * g + r;
            int d  = wn * 32 + ct * 16 + l16;
            if (d < HDIM)
                Out[(size_t)(q0 + rl) * ldo + coff + d] =
                    (_Float16)(acc[ct].f[r] / rowl[rl]);
        }
    }
}

// ---------------------------------------------------------------------------
// y[c][n] = xin[c][n] + t[n][c]   (residual add with transpose)
// ---------------------------------------------------------------------------
__global__ __launch_bounds__(256) void add_residual_kernel(
    const float* __restrict__ xin, const float* __restrict__ tmat,
    float* __restrict__ y) {
    int idx = blockIdx.x * 256 + threadIdx.x;
    if (idx < CCH * NTOK) {
        int c = idx / NTOK, n = idx % NTOK;
        y[idx] = xin[idx] + tmat[(size_t)n * CCH + c];
    }
}

// ---------------------------------------------------------------------------
extern "C" void kernel_launch(void* const* d_in, const int* in_sizes, int n_in,
                              void* d_out, int out_size, void* d_ws, size_t ws_size,
                              hipStream_t stream) {
    (void)in_sizes; (void)n_in; (void)out_size; (void)ws_size;
    const float* x      = (const float*)d_in[0];
    const float* ctx    = (const float*)d_in[1];
    const float* norm_w = (const float*)d_in[2];
    const float* norm_b = (const float*)d_in[3];
    const float* qkv_w  = (const float*)d_in[4];
    const float* qkv_b  = (const float*)d_in[5];
    const float* proj_w = (const float*)d_in[6];
    const float* proj_b = (const float*)d_in[7];
    const float* lnc_w  = (const float*)d_in[8];
    const float* lnc_b  = (const float*)d_in[9];
    const float* qc_w   = (const float*)d_in[10];
    const float* qc_b   = (const float*)d_in[11];
    const float* kvc_w  = (const float*)d_in[12];
    const float* kvc_b  = (const float*)d_in[13];
    const float* pc_w   = (const float*)d_in[14];
    const float* pc_b   = (const float*)d_in[15];
    const float* lnf_w  = (const float*)d_in[16];
    const float* lnf_b  = (const float*)d_in[17];
    const float* f1_w   = (const float*)d_in[18];
    const float* f1_b   = (const float*)d_in[19];
    const float* f2_w   = (const float*)d_in[20];
    const float* f2_b   = (const float*)d_in[21];
    float* out = (float*)d_out;

    // ---- workspace layout: f32 region first, then f16 region -------------
    float* wf = (float*)d_ws;
    size_t o = 0;
    float* gout32 = wf + o; o += (size_t)NTOK * CCH;   // proj / pc / f2 outs
    float* xres   = wf + o; o += (size_t)NTOK * CCH;   // residual stream [C][N]
    _Float16* hh = (_Float16*)(wf + o);
    size_t ho = 0;
    _Float16* tok16  = hh + ho; ho += (size_t)NTOK * CCH;   // gn / ln1 / ln2
    _Float16* qkv16  = hh + ho; ho += (size_t)NTOK * 960;
    _Float16* attn16 = hh + ho; ho += (size_t)NTOK * CCH;
    _Float16* q16    = hh + ho; ho += (size_t)NTOK * CCH;
    _Float16* kv16   = hh + ho; ho += (size_t)80 * 640;
    _Float16* f1buf  = hh + ho; ho += (size_t)NTOK * 1280;
    _Float16* qkvW   = hh + ho; ho += (size_t)960 * 320;
    _Float16* projW  = hh + ho; ho += (size_t)320 * 320;
    _Float16* qcW    = hh + ho; ho += (size_t)320 * 320;
    _Float16* kvcW   = hh + ho; ho += (size_t)640 * 768;
    _Float16* pcW    = hh + ho; ho += (size_t)320 * 320;
    _Float16* f1W    = hh + ho; ho += (size_t)1280 * 320;
    _Float16* f2W    = hh + ho; ho += (size_t)320 * 1280;

    const float scale = 0.15811388f;   // HD^-0.5 = 1/sqrt(40)
    dim3 blk(256);

    // 0) weights f32 -> f16 (once per launch)
    cvt_f16_kernel<<<150, blk, 0, stream>>>(qkv_w, qkvW, 38400);
    cvt_f16_kernel<<<50,  blk, 0, stream>>>(proj_w, projW, 12800);
    cvt_f16_kernel<<<50,  blk, 0, stream>>>(qc_w,  qcW,  12800);
    cvt_f16_kernel<<<240, blk, 0, stream>>>(kvc_w, kvcW, 61440);
    cvt_f16_kernel<<<50,  blk, 0, stream>>>(pc_w,  pcW,  12800);
    cvt_f16_kernel<<<200, blk, 0, stream>>>(f1_w,  f1W,  51200);
    cvt_f16_kernel<<<200, blk, 0, stream>>>(f2_w,  f2W,  51200);
    // context f32 -> f16 staged via tok16? no: kv gemm A = context; convert it
    _Float16* ctx16 = attn16;  // reuse attn16 (77*768 < NTOK*CCH), free now
    cvt_f16_kernel<<<29, blk, 0, stream>>>(ctx, ctx16, 7392);  // 77*768/8

    // 1) GroupNorm -> tok16 [N,C] f16
    groupnorm_kernel<<<32, blk, 0, stream>>>(x, norm_w, norm_b, tok16);
    // 2) qkv = gn @ qkv_w^T -> qkv16 [N,960]
    gemm_kernel<<<dim3(64, 30), blk, 0, stream>>>(tok16, qkvW, qkv_b, nullptr, qkv16, NTOK, 320, 960, 0);
    // 3) self-attention -> attn16 [N,320]
    flash_attn_kernel<<<dim3(64, NHEAD), blk, 0, stream>>>(
        qkv16, 960, qkv16 + 320, 960, qkv16 + 640, 960, attn16, 320, 4096, scale);
    // 4) proj -> gout32; 5) residual: xres = x + proj^T
    gemm_kernel<<<dim3(64, 10), blk, 0, stream>>>(attn16, projW, proj_b, gout32, nullptr, NTOK, 320, 320, 0);
    add_residual_kernel<<<(CCH * NTOK) / 256, blk, 0, stream>>>(x, gout32, xres);
    // 6) LayerNorm -> tok16; 7) q proj -> q16; 8) kv proj (M=77,K=768,O=640)
    layernorm_kernel<<<512, blk, 0, stream>>>(xres, lnc_w, lnc_b, tok16);
    gemm_kernel<<<dim3(64, 10), blk, 0, stream>>>(tok16, qcW, qc_b, nullptr, q16, NTOK, 320, 320, 0);
    gemm_kernel<<<dim3(2, 20), blk, 0, stream>>>(ctx16, kvcW, kvc_b, nullptr, kv16, 77, 768, 640, 0);
    // 9) cross-attention (Lk = 77) -> attn16
    flash_attn_kernel<<<dim3(64, NHEAD), blk, 0, stream>>>(
        q16, 320, kv16, 640, kv16 + 320, 640, attn16, 320, 77, scale);
    // 10) pc proj; 11) residual (in place on xres)
    gemm_kernel<<<dim3(64, 10), blk, 0, stream>>>(attn16, pcW, pc_b, gout32, nullptr, NTOK, 320, 320, 0);
    add_residual_kernel<<<(CCH * NTOK) / 256, blk, 0, stream>>>(xres, gout32, xres);
    // 12) LayerNorm; 13) FFN up + GELU (f16 out); 14) FFN down; 15) residual
    layernorm_kernel<<<512, blk, 0, stream>>>(xres, lnf_w, lnf_b, tok16);
    gemm_kernel<<<dim3(64, 40), blk, 0, stream>>>(tok16, f1W, f1_b, nullptr, f1buf, NTOK, 320, 1280, 1);
    gemm_kernel<<<dim3(64, 10), blk, 0, stream>>>(f1buf, f2W, f2_b, gout32, nullptr, NTOK, 1280, 320, 0);
    add_residual_kernel<<<(CCH * NTOK) / 256, blk, 0, stream>>>(xres, gout32, out);
}
